// NLBlock_56727928046110
// MI455X (gfx1250) — compile-verified, matrix operand order checked by reading, not run on hardware
//
#include <hip/hip_runtime.h>
#include <math.h>

typedef __attribute__((ext_vector_type(16))) _Float16 v16h;
typedef __attribute__((ext_vector_type(8)))  _Float16 v8h;
typedef __attribute__((ext_vector_type(8)))  float    v8f;
typedef __attribute__((ext_vector_type(4)))  int      v4i_;

union F16Frag { v16h v; v8h h[2]; };

#define C_   256
#define CH   128
#define NPOS 6272
#define BATCH 4

#define AS1 __attribute__((address_space(1)))
#define AS3 __attribute__((address_space(3)))

static __device__ __forceinline__ v8f wmma_f16(v16h a, v16h b, v8f c) {
  // D = A(16x32 f16) * B(32x16 f16) + C(16x16 f32)
  return __builtin_amdgcn_wmma_f32_16x16x32_f16(false, a, false, b, (short)0, c,
                                                false, false);
}

// 16-byte global -> LDS async copy (ASYNCcnt) with sync fallback.
static __device__ __forceinline__ void async_cp16(const _Float16* g,
                                                  _Float16* l) {
#if __has_builtin(__builtin_amdgcn_global_load_async_to_lds_b128)
  __builtin_amdgcn_global_load_async_to_lds_b128(
      (AS1 v4i_*)(v4i_*)g, (AS3 v4i_*)(v4i_*)l, 0, 0);
#else
  *(v8h*)l = *(const v8h*)g;
#endif
}

#if __has_builtin(__builtin_amdgcn_s_wait_asynccnt)
#define WAIT_ASYNC(n) __builtin_amdgcn_s_wait_asynccnt(n)
#elif __has_builtin(__builtin_amdgcn_global_load_async_to_lds_b128)
#define WAIT_ASYNC(n) asm volatile("s_wait_asynccnt %0" ::"i"(n) : "memory")
#else
#define WAIT_ASYNC(n) ((void)0)
#endif

// ---------------------------------------------------------------------------
// Kernel 0: BN constants (inv, bias) + convert w_z to f16 [cout][h]
// ---------------------------------------------------------------------------
__global__ void nlb_prep(const float* __restrict__ wz,
                         const float* __restrict__ gamma,
                         const float* __restrict__ beta,
                         const float* __restrict__ rmean,
                         const float* __restrict__ rvar,
                         _Float16* __restrict__ wzT,
                         float* __restrict__ bninv,
                         float* __restrict__ bnbias) {
  int c = threadIdx.x;          // 256 threads
  float iv = gamma[c] * rsqrtf(rvar[c] + 1e-5f);
  bninv[c] = iv;
  bnbias[c] = beta[c] - rmean[c] * iv;
  for (int j = 0; j < CH; ++j)
    wzT[c * CH + j] = (_Float16)wz[c * CH + j];   // w_z is [C][Ch] row-major
}

// ---------------------------------------------------------------------------
// Kernel 1: projections (WMMA f16, f32 accumulate).
//   Q (=theta)  stored [b][n][h]   f16
//   K (=phi^T)  stored [b][n][h]   f16
//   Vt(=g)      stored [b][h][n]   f16
// ---------------------------------------------------------------------------
__global__ __launch_bounds__(256)
void nlb_proj(const float* __restrict__ x,
              const float* __restrict__ wt, const float* __restrict__ wp,
              const float* __restrict__ wg,
              _Float16* __restrict__ Q, _Float16* __restrict__ K,
              _Float16* __restrict__ Vt) {
  __shared__ __attribute__((aligned(16))) _Float16 xs[64 * 264];
  const int b  = blockIdx.y;
  const int nb = blockIdx.x * 64;
  const int tid = threadIdx.x;

  {  // stage x[b][:, nb..nb+63] -> xs[n][c] (f32->f16), coalesced reads
    int n = tid & 63;
    const float* xb = x + (size_t)b * C_ * NPOS + nb + n;
    for (int c = tid >> 6; c < C_; c += 4)
      xs[n * 264 + c] = (_Float16)xb[(size_t)c * NPOS];
  }
  __syncthreads();

  const int lane = tid & 31, wave = tid >> 5;
  const int hi = lane >> 4, col = lane & 15;

  // 96 output tiles (3 mats x 4 n-tiles x 8 h-tiles) / 8 waves = 12 each
  for (int i = 0; i < 12; ++i) {
    int t = wave * 12 + i;
    int mat = t >> 5, r5 = t & 31;
    int nt = r5 >> 3, ht = r5 & 7;
    v8f acc = {};
    if (mat < 2) {
      // D[n,h] = X^T(16x32c from LDS) * W^T(32c x 16h from global f32)
      const float* w = (mat == 0) ? wt : wp;
      const float* wrow = w + (size_t)(ht * 16 + col) * C_;   // B col = h
      #pragma unroll
      for (int kc = 0; kc < 8; ++kc) {
        F16Frag a, bf;
        const _Float16* xp = &xs[(nt * 16 + col) * 264 + kc * 32 + hi * 8];
        a.h[0] = *(const v8h*)xp;
        a.h[1] = *(const v8h*)(xp + 16);
        #pragma unroll
        for (int j = 0; j < 8; ++j) {
          bf.h[0][j] = (_Float16)wrow[kc * 32 + hi * 8 + j];
          bf.h[1][j] = (_Float16)wrow[kc * 32 + 16 + hi * 8 + j];
        }
        acc = wmma_f16(a.v, bf.v, acc);
      }
      _Float16* dst = (mat == 0) ? Q : K;
      #pragma unroll
      for (int r = 0; r < 8; ++r) {
        int m = r + hi * 8;
        dst[((size_t)b * NPOS + nb + nt * 16 + m) * CH + ht * 16 + col] =
            (_Float16)acc[r];
      }
    } else {
      // D[h,n] = W_g(16h x 32c from global) * X(32c x 16n from LDS)
      const float* wrow = wg + (size_t)(ht * 16 + col) * C_;  // A row = h
      #pragma unroll
      for (int kc = 0; kc < 8; ++kc) {
        F16Frag a, bf;
        #pragma unroll
        for (int j = 0; j < 8; ++j) {
          a.h[0][j] = (_Float16)wrow[kc * 32 + hi * 8 + j];
          a.h[1][j] = (_Float16)wrow[kc * 32 + 16 + hi * 8 + j];
        }
        const _Float16* xp = &xs[(nt * 16 + col) * 264 + kc * 32 + hi * 8];
        bf.h[0] = *(const v8h*)xp;
        bf.h[1] = *(const v8h*)(xp + 16);
        acc = wmma_f16(a.v, bf.v, acc);
      }
      #pragma unroll
      for (int r = 0; r < 8; ++r) {
        int m = r + hi * 8;
        Vt[((size_t)b * CH + ht * 16 + m) * NPOS + nb + nt * 16 + col] =
            (_Float16)acc[r];
      }
    }
  }
}

// ---------------------------------------------------------------------------
// Kernel 2: flash attention + w_z projection + BN + residual + ReLU.
// 8 waves/block, 16 query rows per wave.  K/V chunks (32 keys) are staged
// to LDS with double-buffered async global->LDS copies; the post-loop O
// relayout buffer overlays the staging buffers.
//
// LDS carve (halves):
//   kbuf[2] : 2 * 32*136  (key rows padded to 136h -> conflict-free b128)
//   vbuf[2] : 2 * 128*40  (h rows of 32 keys padded to 40h)
//   pbuf    : 8 * 16*32   (per-wave P relayout)
//   obuf    : 8 * 16*136  (post-loop, overlays kbuf/vbuf region)
// ---------------------------------------------------------------------------
#define KB0 0
#define KB1 (32 * 136)
#define VB0 (2 * 32 * 136)
#define VB1 (VB0 + 128 * 40)
#define PB  (VB1 + 128 * 40)
#define SMEM_HALVES (PB + 8 * 16 * 32)

__global__ __launch_bounds__(256)
void nlb_attn(const float* __restrict__ x,
              const _Float16* __restrict__ Q, const _Float16* __restrict__ K,
              const _Float16* __restrict__ Vt, const _Float16* __restrict__ wzT,
              const float* __restrict__ bninv, const float* __restrict__ bnbias,
              float* __restrict__ out) {
  __shared__ __attribute__((aligned(16))) _Float16 smem[SMEM_HALVES];
  const int b = blockIdx.y;
  const int tid = threadIdx.x, lane = tid & 31, wave = tid >> 5;
  const int hi = lane >> 4, col = lane & 15;
  const int qbase = blockIdx.x * 128 + wave * 16;

  const _Float16* Qb = Q + (size_t)b * NPOS * CH;
  const _Float16* Kb = K + (size_t)b * NPOS * CH;
  const _Float16* Vb = Vt + (size_t)b * CH * NPOS;

  F16Frag qf[4];
  #pragma unroll
  for (int c = 0; c < 4; ++c) {     // Q tile resident: 16 rows x 128 ch
    const _Float16* qp = Qb + (size_t)(qbase + col) * CH + c * 32 + hi * 8;
    qf[c].h[0] = *(const v8h*)qp;
    qf[c].h[1] = *(const v8h*)(qp + 16);
  }

  v8f o[8] = {};
  float mrun[8], lrun[8];
  #pragma unroll
  for (int r = 0; r < 8; ++r) { mrun[r] = -3.0e38f; lrun[r] = 0.0f; }

  // staging indices for this thread (4 x b128 per chunk)
  const int krow = tid >> 3, kseg = tid & 7;   // 32 rows x 8 segs of 16h
  const int vrow = tid >> 1, vseg = tid & 1;   // 128 rows x 2 segs of 16h
  const _Float16* kg = Kb + (size_t)krow * CH + kseg * 16;
  const _Float16* vg = Vb + (size_t)vrow * NPOS + vseg * 16;
  const int kl = krow * 136 + kseg * 16;
  const int vl = vrow * 40 + vseg * 16;

  // prefetch chunk 0
  async_cp16(kg, &smem[KB0 + kl]);
  async_cp16(kg + 8, &smem[KB0 + kl + 8]);
  async_cp16(vg, &smem[VB0 + vl]);
  async_cp16(vg + 8, &smem[VB0 + vl + 8]);

  _Float16* pb = &smem[PB + wave * 512];
  const int nchunk = NPOS / 32;
  for (int it = 0; it < nchunk; ++it) {
    if (it + 1 < nchunk) {      // prefetch next chunk into other buffer
      int nkb = (it + 1) * 32;
      int nb_ = (it + 1) & 1;
      const _Float16* kgn = kg + (size_t)nkb * CH;
      const _Float16* vgn = vg + nkb;
      _Float16* kln = &smem[(nb_ ? KB1 : KB0) + kl];
      _Float16* vln = &smem[(nb_ ? VB1 : VB0) + vl];
      async_cp16(kgn, kln);
      async_cp16(kgn + 8, kln + 8);
      async_cp16(vgn, vln);
      async_cp16(vgn + 8, vln + 8);
      WAIT_ASYNC(4);            // chunk `it` has landed
    } else {
      WAIT_ASYNC(0);
    }
    __syncthreads();
    const _Float16* kc = &smem[(it & 1) ? KB1 : KB0];
    const _Float16* vc = &smem[(it & 1) ? VB1 : VB0];

    // ---- S = Q * K^T  (16 x 32 scores, two 16x16 f32 tiles) ----
    v8f s0 = {}, s1 = {};
    #pragma unroll
    for (int c = 0; c < 4; ++c) {
      F16Frag b0, b1;
      const _Float16* kp0 = kc + col * 136 + c * 32 + hi * 8;
      b0.h[0] = *(const v8h*)kp0;
      b0.h[1] = *(const v8h*)(kp0 + 16);
      const _Float16* kp1 = kp0 + 16 * 136;
      b1.h[0] = *(const v8h*)kp1;
      b1.h[1] = *(const v8h*)(kp1 + 16);
      s0 = wmma_f16(qf[c].v, b0.v, s0);
      s1 = wmma_f16(qf[c].v, b1.v, s1);
    }
    // ---- online softmax over this 32-key chunk ----
    #pragma unroll
    for (int r = 0; r < 8; ++r) {
      float v0 = s0[r], v1 = s1[r];
      float mx = fmaxf(v0, v1);
      #pragma unroll
      for (int off = 1; off < 16; off <<= 1)
        mx = fmaxf(mx, __shfl_xor(mx, off));        // 16-lane row reduce
      float mnew  = fmaxf(mrun[r], mx);
      float alpha = __expf(mrun[r] - mnew);
      float p0 = __expf(v0 - mnew);
      float p1 = __expf(v1 - mnew);
      float rs = p0 + p1;
      #pragma unroll
      for (int off = 1; off < 16; off <<= 1)
        rs += __shfl_xor(rs, off);
      lrun[r] = lrun[r] * alpha + rs;
      mrun[r] = mnew;
      #pragma unroll
      for (int t = 0; t < 8; ++t) o[t][r] *= alpha;
      int m = r + hi * 8;
      pb[m * 32 + col]      = (_Float16)p0;         // C-layout -> LDS [m][k]
      pb[m * 32 + 16 + col] = (_Float16)p1;
    }
    __builtin_amdgcn_wave_barrier();   // wave-private LDS; DS is in-order
    // ---- P (16x32) as A-fragment, O += P * V ----
    F16Frag pa;
    pa.h[0] = *(const v8h*)&pb[col * 32 + hi * 8];
    pa.h[1] = *(const v8h*)&pb[col * 32 + 16 + hi * 8];
    #pragma unroll
    for (int t = 0; t < 8; ++t) {
      F16Frag vb;
      const _Float16* vp = vc + (t * 16 + col) * 40 + hi * 8;
      vb.h[0] = *(const v8h*)vp;
      vb.h[1] = *(const v8h*)(vp + 16);
      o[t] = wmma_f16(pa.v, vb.v, o[t]);
    }
    __syncthreads();   // all waves done with buffer before it is re-filled
  }

  // ---- normalize, relayout O through LDS (overlays staging bufs) ----
  #pragma unroll
  for (int r = 0; r < 8; ++r) {
    float invl = 1.0f / lrun[r];
    #pragma unroll
    for (int t = 0; t < 8; ++t) o[t][r] *= invl;
  }
  _Float16* ob = &smem[wave * 2176];   // 16*136 per wave
  #pragma unroll
  for (int t = 0; t < 8; ++t)
    #pragma unroll
    for (int r = 0; r < 8; ++r)
      ob[(r + hi * 8) * 136 + t * 16 + col] = (_Float16)o[t][r];
  __builtin_amdgcn_wave_barrier();
  F16Frag of[4];
  #pragma unroll
  for (int c = 0; c < 4; ++c) {
    of[c].h[0] = *(const v8h*)&ob[col * 136 + c * 32 + hi * 8];
    of[c].h[1] = *(const v8h*)&ob[col * 136 + c * 32 + 16 + hi * 8];
  }

  // ---- Z = O * w_z^T, fused BN + residual + ReLU ----
  const float* xb  = x   + (size_t)b * C_ * NPOS;
  float*       oub = out + (size_t)b * C_ * NPOS;
  for (int ct = 0; ct < 16; ++ct) {
    v8f z = {};
    #pragma unroll
    for (int c = 0; c < 4; ++c) {
      F16Frag wb;
      const _Float16* wp = wzT + (size_t)(ct * 16 + col) * CH + c * 32 + hi * 8;
      wb.h[0] = *(const v8h*)wp;
      wb.h[1] = *(const v8h*)(wp + 16);
      z = wmma_f16(of[c].v, wb.v, z);
    }
    int cout = ct * 16 + col;
    float iv = bninv[cout], bs = bnbias[cout];
    #pragma unroll
    for (int r = 0; r < 8; ++r) {
      int n = qbase + r + hi * 8;
      size_t idx = (size_t)cout * NPOS + n;
      float val = z[r] * iv + bs + xb[idx];
      oub[idx] = val > 0.0f ? val : 0.0f;
    }
  }
}

// ---------------------------------------------------------------------------
extern "C" void kernel_launch(void* const* d_in, const int* in_sizes, int n_in,
                              void* d_out, int out_size, void* d_ws,
                              size_t ws_size, hipStream_t stream) {
  const float* x     = (const float*)d_in[0];
  const float* wt    = (const float*)d_in[1];
  const float* wp    = (const float*)d_in[2];
  const float* wg    = (const float*)d_in[3];
  const float* wz    = (const float*)d_in[4];
  const float* gamma = (const float*)d_in[5];
  const float* beta  = (const float*)d_in[6];
  const float* rmean = (const float*)d_in[7];
  const float* rvar  = (const float*)d_in[8];
  float* out = (float*)d_out;

  char* ws = (char*)d_ws;
  const size_t szQKV = (size_t)BATCH * NPOS * CH * sizeof(_Float16); // ~6.4 MB
  _Float16* Q   = (_Float16*)(ws);
  _Float16* K   = (_Float16*)(ws + szQKV);
  _Float16* Vt  = (_Float16*)(ws + 2 * szQKV);
  _Float16* wzT = (_Float16*)(ws + 3 * szQKV);
  float* bninv  = (float*)(ws + 3 * szQKV + (size_t)C_ * CH * sizeof(_Float16));
  float* bnbias = bninv + C_;

  nlb_prep<<<1, 256, 0, stream>>>(wz, gamma, beta, rmean, rvar, wzT, bninv,
                                  bnbias);
  nlb_proj<<<dim3(NPOS / 64, BATCH), 256, 0, stream>>>(x, wt, wp, wg, Q, K, Vt);
  nlb_attn<<<dim3(NPOS / 128, BATCH), 256, 0, stream>>>(x, Q, K, Vt, wzT, bninv,
                                                        bnbias, out);
}
